// HybridConvLayer_17987323036195
// MI455X (gfx1250) — matrix operand order repaired and checked by reading, not compiled, not run on hardware
//
#include <hip/hip_runtime.h>
#include <hip/hip_bf16.h>
#include <cmath>
#include <cstdint>

#define D_DIM 128

typedef __bf16 bf16_t;
typedef __attribute__((ext_vector_type(16))) __bf16 bf16x16;
typedef __attribute__((ext_vector_type(8)))  __bf16 bf16x8;
typedef __attribute__((ext_vector_type(8)))  float  f32x8;

__device__ __forceinline__ float gelu_f(float v) {
    // exact erf-based GELU (matches torch nn.GELU default)
    return 0.5f * v * (1.0f + erff(v * 0.70710678118654752440f));
}

__device__ __forceinline__ f32x8 wmma_bf16(bf16x16 a, bf16x16 b, f32x8 c) {
    // D = A(16x32 bf16) * B(32x16 bf16) + C(16x16 f32)
    return __builtin_amdgcn_wmma_f32_16x16x32_bf16(false, a, false, b, (short)0, c,
                                                   false, false);
}

__device__ __forceinline__ bf16x16 cat8(bf16x8 lo, bf16x8 hi) {
    return __builtin_shufflevector(lo, hi, 0, 1, 2, 3, 4, 5, 6, 7,
                                           8, 9, 10, 11, 12, 13, 14, 15);
}

// ---- A loader from pre-converted bf16 row-major feats: 16 rows x 32 K ----
// ISA 16-bit A 16x32 layout: lanes 0-15 row M=lane, elems 0..7 = K k0..k0+7,
// elems 8..15 = K k0+16..k0+23; lanes 16-31 row M=lane-16, K +8 / +24.
// -> two contiguous 16-byte loads per lane.
__device__ __forceinline__ bf16x16 load_a_bf16(const bf16_t* __restrict__ s,
                                               int r0, int k0, int lane) {
    const int half = (lane >> 4) & 1;
    const bf16_t* p = s + (size_t)(r0 + (lane & 15)) * D_DIM + k0 + half * 8;
    bf16x8 lo = *(const bf16x8*)(p);
    bf16x8 hi = *(const bf16x8*)(p + 16);
    return cat8(lo, hi);
}

// ---- A loader from fp32 source (edge_feats: single-use, convert in-register) ----
__device__ __forceinline__ bf16x16 load_a_f32(const float* __restrict__ f,
                                              int r0, int k0, int lane) {
    const int half = (lane >> 4) & 1;
    const float* p = f + (size_t)(r0 + (lane & 15)) * D_DIM + k0 + half * 8;
    bf16x16 a;
#pragma unroll
    for (int i = 0; i < 8; ++i) {
        a[i]     = (__bf16)p[i];
        a[i + 8] = (__bf16)p[16 + i];
    }
    return a;
}

// ---- B loader from pre-packed bf16 weights: one 32-byte contiguous read ----
// packed index: (((kt*8 + nt)*32 + lane)*16 + i)
__device__ __forceinline__ bf16x16 load_b_packed(const bf16_t* __restrict__ pw,
                                                 int kt, int nt, int lane) {
    const bf16_t* p = pw + ((size_t)(kt * 8 + nt) * 32 + lane) * 16;
    return *(const bf16x16*)p;
}

// ======================= weight pre-pack =======================
// From row-major fp32 [K x 128] into per-lane WMMA-B bf16 layout:
// lanes 0-15 col n0+lane, K k0..k0+7 then k0+16..k0+23; lanes 16-31 K +8 / +24.
__global__ __launch_bounds__(256) void pack_b_kernel(const float* __restrict__ w,
                                                     bf16_t* __restrict__ out, int K) {
    int t = blockIdx.x * blockDim.x + threadIdx.x;
    int total = (K >> 5) * 8 * 32;
    if (t >= total) return;
    int lane = t & 31;
    int nt = (t >> 5) & 7;
    int kt = t >> 8;
    int half = (lane >> 4) & 1;
    int col = nt * 16 + (lane & 15);
    const float* p = w + (size_t)(kt * 32 + half * 8) * D_DIM + col;
    bf16_t* o = out + (size_t)t * 16;
#pragma unroll
    for (int i = 0; i < 16; ++i) o[i] = (__bf16)p[(size_t)(i + (i & 8)) * D_DIM];
}

// ======================= feature bf16 conversion =======================
// fx..fx4 = bf16(x..x4); fd0..fd2 = bf16 band diffs
__global__ __launch_bounds__(256) void cvt_feats_kernel(
    const float* __restrict__ x, const float* __restrict__ x1,
    const float* __restrict__ x2, const float* __restrict__ x4,
    bf16_t* __restrict__ fx, bf16_t* __restrict__ fx1, bf16_t* __restrict__ fx2,
    bf16_t* __restrict__ fx4, bf16_t* __restrict__ fd0, bf16_t* __restrict__ fd1,
    bf16_t* __restrict__ fd2, int total) {
    int i = blockIdx.x * blockDim.x + threadIdx.x;
    if (i >= total) return;
    float a = x[i], b = x1[i], c = x2[i], d = x4[i];
    fx[i]  = (__bf16)a;  fx1[i] = (__bf16)b;
    fx2[i] = (__bf16)c;  fx4[i] = (__bf16)d;
    fd0[i] = (__bf16)(a - b);
    fd1[i] = (__bf16)(b - c);
    fd2[i] = (__bf16)(c - d);
}

// fxc = bf16(x4 + out_low + out_band)
__global__ __launch_bounds__(256) void xc_kernel(const float* __restrict__ x4,
                                                 const float* __restrict__ ol,
                                                 const float* __restrict__ ob,
                                                 bf16_t* __restrict__ fxc, int total) {
    int i = blockIdx.x * blockDim.x + threadIdx.x;
    if (i >= total) return;
    fxc[i] = (__bf16)(x4[i] + ol[i] + ob[i]);
}

// ======================= 1) Edge-weight MLP =======================
// ew[e] = sigmoid( gelu(ef @ w1 + b1) @ w2 + b2 ), one wave per 16 edges
__global__ __launch_bounds__(256) void edge_mlp_kernel(
    const float* __restrict__ ef, const bf16_t* __restrict__ pw1,
    const float* __restrict__ b1, const float* __restrict__ w2,
    const float* __restrict__ b2, float* __restrict__ ew, int E) {
    const int lane = threadIdx.x & 31;
    const int wave = blockIdx.x * (blockDim.x >> 5) + (threadIdx.x >> 5);
    const int r0 = wave * 16;
    if (r0 >= E) return;

    bf16x16 a[4];
#pragma unroll
    for (int k = 0; k < 4; ++k) a[k] = load_a_f32(ef, r0, k * 32, lane);

    float dot[8] = {};
    const float b2s = b2[0];
#pragma unroll
    for (int nt = 0; nt < 8; ++nt) {
        f32x8 acc = {};
#pragma unroll
        for (int k = 0; k < 4; ++k)
            acc = wmma_bf16(a[k], load_b_packed(pw1, k, nt, lane), acc);
        const int col = nt * 16 + (lane & 15);
        const float b1c = b1[col], w2c = w2[col];
#pragma unroll
        for (int v = 0; v < 8; ++v) dot[v] += gelu_f(acc[v] + b1c) * w2c;
    }
    // reduce over 128 columns: butterfly within each 16-lane half
#pragma unroll
    for (int v = 0; v < 8; ++v) {
#pragma unroll
        for (int off = 1; off < 16; off <<= 1) dot[v] += __shfl_xor(dot[v], off, 16);
    }
    if ((lane & 15) == 0) {
        const int rbase = r0 + ((lane >> 4) << 3);
#pragma unroll
        for (int v = 0; v < 8; ++v)
            ew[rbase + v] = 1.0f / (1.0f + expf(-(dot[v] + b2s)));
    }
}

// ======================= 2) Degrees / dinv =======================
__global__ __launch_bounds__(256) void deg_init_kernel(float* __restrict__ deg, int N) {
    int i = blockIdx.x * blockDim.x + threadIdx.x;
    if (i < N) deg[i] = 1.0f;  // self loop weight
}

__global__ __launch_bounds__(256) void deg_acc_kernel(const int* __restrict__ col,
                                                      const float* __restrict__ ew,
                                                      float* __restrict__ deg, int E) {
    int e = blockIdx.x * blockDim.x + threadIdx.x;
    if (e < E) unsafeAtomicAdd(&deg[col[e]], ew[e]);
}

__global__ __launch_bounds__(256) void dinv_kernel(const float* __restrict__ deg,
                                                   float* __restrict__ dinv, int N) {
    int i = blockIdx.x * blockDim.x + threadIdx.x;
    if (i < N) {
        float d = deg[i];
        dinv[i] = (d > 0.0f) ? rsqrtf(d) : 0.0f;
    }
}

// ======================= 3) Propagation =======================
__global__ __launch_bounds__(256) void prop_self_kernel(const float* __restrict__ h,
                                                        const float* __restrict__ dinv,
                                                        float* __restrict__ out, int N) {
    int i = blockIdx.x * blockDim.x + threadIdx.x;
    if (i >= N * D_DIM) return;
    float d = dinv[i >> 7];
    out[i] = d * d * h[i];
}

// one thread per (edge, 4 consecutive features): coalesced gather + HW f32 atomics
__global__ __launch_bounds__(256) void prop_edge_kernel(
    const float* __restrict__ h, const int* __restrict__ row,
    const int* __restrict__ col, const float* __restrict__ ew,
    const float* __restrict__ dinv, float* __restrict__ out, int E) {
    long long t = (long long)blockIdx.x * blockDim.x + threadIdx.x;
    int e = (int)(t >> 5);
    if (e >= E) return;
    int seg = ((int)t & 31) << 2;
    int r = row[e], c = col[e];
    float wgt = dinv[r] * ew[e] * dinv[c];
    const float4 hv = *reinterpret_cast<const float4*>(h + (size_t)r * D_DIM + seg);
    float* o = out + (size_t)c * D_DIM + seg;
    unsafeAtomicAdd(o + 0, wgt * hv.x);
    unsafeAtomicAdd(o + 1, wgt * hv.y);
    unsafeAtomicAdd(o + 2, wgt * hv.z);
    unsafeAtomicAdd(o + 3, wgt * hv.w);
}

// ======================= 4) pre = gelu(x); p = pre @ a_pre^T =======================
__global__ __launch_bounds__(256) void pre_p_kernel(const float* __restrict__ x,
                                                    const float* __restrict__ apre_l,
                                                    const float* __restrict__ apre_b,
                                                    float* __restrict__ pl,
                                                    float* __restrict__ pb, int N) {
    const int lane = threadIdx.x & 31;
    const int node = blockIdx.x * (blockDim.x >> 5) + (threadIdx.x >> 5);
    if (node >= N) return;
    float g[4];
#pragma unroll
    for (int j = 0; j < 4; ++j) g[j] = gelu_f(x[(size_t)node * D_DIM + lane + 32 * j]);
#pragma unroll
    for (int h = 0; h < 2; ++h) {
        float s1 = 0.f, s2 = 0.f;
#pragma unroll
        for (int j = 0; j < 4; ++j) {
            int d = lane + 32 * j;
            s1 += g[j] * apre_l[h * D_DIM + d];
            s2 += g[j] * apre_b[h * D_DIM + d];
        }
#pragma unroll
        for (int off = 1; off < 32; off <<= 1) {
            s1 += __shfl_xor(s1, off, 32);
            s2 += __shfl_xor(s2, off, 32);
        }
        if (lane == 0) { pl[node * 2 + h] = s1; pb[node * 2 + h] = s2; }
    }
}

// ======================= 5) Attention scores (per group) =======================
// one wave per (16-node tile, channel, head): z = feat_c @ W[c,h] + b, reduce to score
__global__ __launch_bounds__(256) void score_kernel(
    const bf16_t* f0, const bf16_t* f1, const bf16_t* f2, const bf16_t* f3,
    const bf16_t* __restrict__ pWg, const float* __restrict__ bg,
    const float* __restrict__ attch, const float* __restrict__ p,
    float* __restrict__ scores, int N, int C) {
    const int lane = threadIdx.x & 31;
    const int wave = blockIdx.x * (blockDim.x >> 5) + (threadIdx.x >> 5);
    const int tiles = (N + 15) >> 4;
    const int per = C * 2;
    const int tile = wave / per;
    if (tile >= tiles) return;
    const int rem = wave % per;
    const int c = rem >> 1, h = rem & 1;
    const int r0 = tile * 16;
    const bf16_t* fp[4] = {f0, f1, f2, f3};

    bf16x16 a[4];
#pragma unroll
    for (int k = 0; k < 4; ++k) a[k] = load_a_bf16(fp[c], r0, k * 32, lane);

    const bf16_t* W   = pWg + (size_t)(c * 2 + h) * D_DIM * D_DIM;
    const float* bias = bg + (size_t)(c * 2 + h) * D_DIM;
    float part[8] = {};
#pragma unroll
    for (int nt = 0; nt < 8; ++nt) {
        f32x8 acc = {};
#pragma unroll
        for (int k = 0; k < 4; ++k)
            acc = wmma_bf16(a[k], load_b_packed(W, k, nt, lane), acc);
        const int col = nt * 16 + (lane & 15);
        const float bcol = bias[col], av = attch[h * D_DIM + col];
#pragma unroll
        for (int v = 0; v < 8; ++v) part[v] += gelu_f(acc[v] + bcol) * av;
    }
#pragma unroll
    for (int v = 0; v < 8; ++v) {
#pragma unroll
        for (int off = 1; off < 16; off <<= 1) part[v] += __shfl_xor(part[v], off, 16);
    }
    if ((lane & 15) == 0) {
        const int rbase = r0 + ((lane >> 4) << 3);
#pragma unroll
        for (int v = 0; v < 8; ++v) {
            const int n = rbase + v;
            scores[((size_t)c * N + n) * 2 + h] = gelu_f(p[n * 2 + h] + part[v]);
        }
    }
}

// ======================= 6) softmax over channels (in place) =======================
__global__ __launch_bounds__(256) void softmax_kernel(float* __restrict__ s, int N, int C) {
    int t = blockIdx.x * blockDim.x + threadIdx.x;
    if (t >= N * 2) return;
    int n = t >> 1, h = t & 1;
    float v[4];
    float m = -3.4e38f;
    for (int c = 0; c < C; ++c) {
        v[c] = s[((size_t)c * N + n) * 2 + h];
        m = fmaxf(m, v[c]);
    }
    float sum = 0.f;
    for (int c = 0; c < C; ++c) { v[c] = expf(v[c] - m); sum += v[c]; }
    float inv = 1.f / sum;
    for (int c = 0; c < C; ++c) s[((size_t)c * N + n) * 2 + h] = v[c] * inv;
}

// ======================= 7) Combine (recompute z, weight by alpha, mean heads) ====
__global__ __launch_bounds__(256) void combine_kernel(
    const bf16_t* f0, const bf16_t* f1, const bf16_t* f2, const bf16_t* f3,
    const bf16_t* __restrict__ pWg, const float* __restrict__ bg,
    const float* __restrict__ alpha, float* __restrict__ outg, int N, int C) {
    const int lane = threadIdx.x & 31;
    const int tile = blockIdx.x * (blockDim.x >> 5) + (threadIdx.x >> 5);
    if (tile * 16 >= N) return;
    const int r0 = tile * 16;
    const int rbase = r0 + ((lane >> 4) << 3);
    const bf16_t* fp[4] = {f0, f1, f2, f3};

    f32x8 o[8] = {};
    for (int c = 0; c < C; ++c) {
        bf16x16 a[4];
#pragma unroll
        for (int k = 0; k < 4; ++k) a[k] = load_a_bf16(fp[c], r0, k * 32, lane);
#pragma unroll
        for (int h = 0; h < 2; ++h) {
            const bf16_t* W   = pWg + (size_t)(c * 2 + h) * D_DIM * D_DIM;
            const float* bias = bg + (size_t)(c * 2 + h) * D_DIM;
            float al[8];
#pragma unroll
            for (int v = 0; v < 8; ++v)
                al[v] = alpha[((size_t)c * N + rbase + v) * 2 + h];
#pragma unroll
            for (int nt = 0; nt < 8; ++nt) {
                f32x8 acc = {};
#pragma unroll
                for (int k = 0; k < 4; ++k)
                    acc = wmma_bf16(a[k], load_b_packed(W, k, nt, lane), acc);
                const float bcol = bias[nt * 16 + (lane & 15)];
#pragma unroll
                for (int v = 0; v < 8; ++v) o[nt][v] += al[v] * (acc[v] + bcol);
            }
        }
    }
#pragma unroll
    for (int nt = 0; nt < 8; ++nt) {
        const int col = nt * 16 + (lane & 15);
#pragma unroll
        for (int v = 0; v < 8; ++v)
            outg[(size_t)(rbase + v) * D_DIM + col] = o[nt][v] * 0.5f;  // mean over H=2
    }
}

// ======================= 8) Final fused: [x | xc] @ w_out + b_out =================
__global__ __launch_bounds__(256) void final_kernel(
    const bf16_t* __restrict__ fx, const bf16_t* __restrict__ fxc,
    const bf16_t* __restrict__ pwo, const float* __restrict__ bo,
    float* __restrict__ out, int N) {
    const int lane = threadIdx.x & 31;
    const int tile = blockIdx.x * (blockDim.x >> 5) + (threadIdx.x >> 5);
    if (tile * 16 >= N) return;
    const int r0 = tile * 16;
    const int rbase = r0 + ((lane >> 4) << 3);
    bf16x16 ax[4], ac[4];
#pragma unroll
    for (int k = 0; k < 4; ++k) {
        ax[k] = load_a_bf16(fx, r0, k * 32, lane);
        ac[k] = load_a_bf16(fxc, r0, k * 32, lane);
    }
#pragma unroll
    for (int nt = 0; nt < 8; ++nt) {
        f32x8 acc = {};
#pragma unroll
        for (int k = 0; k < 4; ++k)
            acc = wmma_bf16(ax[k], load_b_packed(pwo, k, nt, lane), acc);
#pragma unroll
        for (int k = 0; k < 4; ++k)
            acc = wmma_bf16(ac[k], load_b_packed(pwo, 4 + k, nt, lane), acc);
        const int col = nt * 16 + (lane & 15);
        const float bc = bo[col];
#pragma unroll
        for (int v = 0; v < 8; ++v)
            out[(size_t)(rbase + v) * D_DIM + col] = acc[v] + bc;
    }
}

// ======================= host side =======================
extern "C" void kernel_launch(void* const* d_in, const int* in_sizes, int n_in,
                              void* d_out, int out_size, void* d_ws, size_t ws_size,
                              hipStream_t stream) {
    const float* x    = (const float*)d_in[0];
    const int*   eidx = (const int*)d_in[1];
    const float* ef   = (const float*)d_in[2];
    const float* w1   = (const float*)d_in[3];
    const float* b1   = (const float*)d_in[4];
    const float* w2   = (const float*)d_in[5];
    const float* b2   = (const float*)d_in[6];
    const float* Wh   = (const float*)d_in[7];
    const float* bh   = (const float*)d_in[8];
    const float* apl  = (const float*)d_in[9];
    const float* acl  = (const float*)d_in[10];
    const float* apb  = (const float*)d_in[11];
    const float* acb  = (const float*)d_in[12];
    const float* wo   = (const float*)d_in[13];
    const float* bo   = (const float*)d_in[14];
    float* out = (float*)d_out;

    const int N = in_sizes[0] / D_DIM;
    const int E = in_sizes[2] / D_DIM;
    const int* row = eidx;      // edge_index[0] (source)
    const int* col = eidx + E;  // edge_index[1] (destination)
    const size_t ND = (size_t)N * D_DIM;

    // ---- workspace carve-up: fp32 region, then 256B-aligned bf16 region ----
    float* ws = (float*)d_ws;
    float* ew   = ws; ws += E;
    float* deg  = ws; ws += N;
    float* dinv = ws; ws += N;
    float* x1 = ws; ws += ND;
    float* x2 = ws; ws += ND;
    float* x3 = ws; ws += ND;
    float* x4 = ws; ws += ND;
    float* pl = ws; ws += (size_t)N * 2;
    float* pb = ws; ws += (size_t)N * 2;
    float* sl = ws; ws += (size_t)4 * N * 2;
    float* sb = ws; ws += (size_t)3 * N * 2;
    float* ol = ws; ws += ND;
    float* ob = ws; ws += ND;
    uintptr_t up = ((uintptr_t)ws + 255) & ~(uintptr_t)255;
    bf16_t* bptr = (bf16_t*)up;
    bf16_t* fx  = bptr; bptr += ND;   // bf16 feats
    bf16_t* fx1 = bptr; bptr += ND;
    bf16_t* fx2 = bptr; bptr += ND;
    bf16_t* fx4 = bptr; bptr += ND;
    bf16_t* fd0 = bptr; bptr += ND;
    bf16_t* fd1 = bptr; bptr += ND;
    bf16_t* fd2 = bptr; bptr += ND;
    bf16_t* fxc = bptr; bptr += ND;
    bf16_t* pw1 = bptr; bptr += (size_t)D_DIM * D_DIM;       // packed weights
    bf16_t* pwh = bptr; bptr += (size_t)14 * D_DIM * D_DIM;
    bf16_t* pwo = bptr; bptr += (size_t)2 * D_DIM * D_DIM;
    (void)ws_size; (void)n_in; (void)out_size;

    const int TPB = 256;                 // 8 waves per block (wave32)
    const int tiles = (N + 15) / 16;

    // 0) pack all GEMM weights once into bf16 WMMA-B layout
    pack_b_kernel<<<4, TPB, 0, stream>>>(w1, pw1, D_DIM);
    for (int m = 0; m < 14; ++m)
        pack_b_kernel<<<4, TPB, 0, stream>>>(Wh + (size_t)m * D_DIM * D_DIM,
                                             pwh + (size_t)m * D_DIM * D_DIM, D_DIM);
    pack_b_kernel<<<8, TPB, 0, stream>>>(wo, pwo, 2 * D_DIM);

    // 1) edge MLP -> ew
    {
        int waves = (E + 15) / 16;
        edge_mlp_kernel<<<(waves + 7) / 8, TPB, 0, stream>>>(ef, pw1, b1, w2, b2, ew, E);
    }
    // 2) degrees & dinv
    deg_init_kernel<<<(N + TPB - 1) / TPB, TPB, 0, stream>>>(deg, N);
    deg_acc_kernel<<<(E + TPB - 1) / TPB, TPB, 0, stream>>>(col, ew, deg, E);
    dinv_kernel<<<(N + TPB - 1) / TPB, TPB, 0, stream>>>(deg, dinv, N);
    // 3) four propagation rounds: x1=P x, x2=P x1, x3=P x2, x4=P x3
    {
        const float* src = x;
        float* dsts[4] = {x1, x2, x3, x4};
        const long long edge_threads = (long long)E * 32;
        for (int i = 0; i < 4; ++i) {
            prop_self_kernel<<<(N * D_DIM + TPB - 1) / TPB, TPB, 0, stream>>>(src, dinv,
                                                                              dsts[i], N);
            prop_edge_kernel<<<(unsigned)((edge_threads + TPB - 1) / TPB), TPB, 0,
                               stream>>>(src, row, col, ew, dinv, dsts[i], E);
            src = dsts[i];
        }
    }
    // 4) bf16 channel feats + pre/p
    cvt_feats_kernel<<<(int)((ND + TPB - 1) / TPB), TPB, 0, stream>>>(
        x, x1, x2, x4, fx, fx1, fx2, fx4, fd0, fd1, fd2, (int)ND);
    pre_p_kernel<<<(N + 7) / 8, TPB, 0, stream>>>(x, apl, apb, pl, pb, N);
    // 5) scores: low group (x,x1,x2,x4), band group (x-x1, x1-x2, x2-x4)
    {
        int waves = tiles * 4 * 2;
        score_kernel<<<(waves + 7) / 8, TPB, 0, stream>>>(
            fx, fx1, fx2, fx4, pwh, bh, acl, pl, sl, N, 4);
    }
    {
        int waves = tiles * 3 * 2;
        score_kernel<<<(waves + 7) / 8, TPB, 0, stream>>>(
            fd0, fd1, fd2, nullptr, pwh + (size_t)8 * D_DIM * D_DIM,
            bh + (size_t)8 * D_DIM, acb, pb, sb, N, 3);
    }
    // 6) softmax over channels (in place -> alpha)
    softmax_kernel<<<(N * 2 + TPB - 1) / TPB, TPB, 0, stream>>>(sl, N, 4);
    softmax_kernel<<<(N * 2 + TPB - 1) / TPB, TPB, 0, stream>>>(sb, N, 3);
    // 7) combine groups
    combine_kernel<<<(tiles + 7) / 8, TPB, 0, stream>>>(
        fx, fx1, fx2, fx4, pwh, bh, sl, ol, N, 4);
    combine_kernel<<<(tiles + 7) / 8, TPB, 0, stream>>>(
        fd0, fd1, fd2, nullptr, pwh + (size_t)8 * D_DIM * D_DIM,
        bh + (size_t)8 * D_DIM, sb, ob, N, 3);
    // 8) xc in bf16, then final fused linear
    xc_kernel<<<(int)((ND + TPB - 1) / TPB), TPB, 0, stream>>>(x4, ol, ob, fxc, (int)ND);
    final_kernel<<<(tiles + 7) / 8, TPB, 0, stream>>>(fx, fxc, pwo, bo, out, N);
}
